// DTR_6803228197468
// MI455X (gfx1250) — matrix operand order repaired and checked by reading, not compiled, not run on hardware
//
#include <hip/hip_runtime.h>
#include <hip/hip_bf16.h>

// ---------------------------------------------------------------------------
// Problem constants (match reference)
// ---------------------------------------------------------------------------
#define BC   128        // clusters
#define RR   16         // reads per cluster
#define SS   (BC*RR)    // 2048 sequences
#define LL   128        // read length
#define WW   9          // window
#define CC   64         // emb channels
#define DD   256        // lstm dim
#define NHH  4          // heads

typedef __bf16 bf16;
typedef __bf16 v16bf __attribute__((ext_vector_type(16)));
typedef float  v8f   __attribute__((ext_vector_type(8)));
typedef unsigned int u32x4 __attribute__((ext_vector_type(4)));

union FragU { v16bf v; u32x4 q[2]; };

// A-fragment (16x32 bf16): lanes 0-15 hold rows M=0..15; hi half-lane group
// takes K offsets +8 (first 8 VGPR halves) / +24 (second). Two 16B loads.
__device__ __forceinline__ v16bf load_a_frag(const bf16* __restrict__ p, int hi) {
  FragU f;
  const bf16* b = p + hi * 8;
  f.q[0] = *reinterpret_cast<const u32x4*>(b);
  f.q[1] = *reinterpret_cast<const u32x4*>(b + 16);
  return f.v;
}

// B-fragment (32x16 bf16), weights stored row-major (N,K): lane n holds column
// N=n; hi half-lane group takes K offsets +8 / +24, matching the A-fragment.
__device__ __forceinline__ v16bf load_b_frag(const bf16* __restrict__ p, int hi) {
  FragU f;
  const bf16* b = p + hi * 8;
  f.q[0] = *reinterpret_cast<const u32x4*>(b);
  f.q[1] = *reinterpret_cast<const u32x4*>(b + 16);
  return f.v;
}

// ---------------------------------------------------------------------------
// WMMA GEMM:  out[m][n] = act( A[m][:] . B[n][:] + bias[n] )
//   A: (M,K) bf16 row-major, row stride lda
//   B: (N,K) bf16 row-major (i.e. torch-style weight), row stride ldb
//   Cf: optional f32 output (stride ldcf); Cb: optional bf16 output (ldcb)
// Block: 256 threads = 8 waves; wave tile 32x32 (2x2 wmma); block tile 128x64.
// Requires M%128==0 (or M==128 with grid.y==1), N%64==0, K%32==0.
// ---------------------------------------------------------------------------
__global__ __launch_bounds__(256)
void gemm_bf16_wmma(const bf16* __restrict__ A, int lda,
                    const bf16* __restrict__ B, int ldb,
                    const float* __restrict__ bias,
                    float* __restrict__ Cf, int ldcf,
                    bf16* __restrict__ Cb, int ldcb,
                    int K, int relu)
{
  const int tid    = threadIdx.x;
  const int lane   = tid & 31;
  const int laneLo = lane & 15;
  const int hi     = lane >> 4;
  const int waveId = tid >> 5;
  const int m0 = blockIdx.y * 128 + (waveId >> 1) * 32;
  const int n0 = blockIdx.x * 64  + (waveId & 1)  * 32;

  v8f acc00 = {}, acc01 = {}, acc10 = {}, acc11 = {};

  const bf16* arow0 = A + (size_t)(m0 + laneLo)      * lda;
  const bf16* arow1 = A + (size_t)(m0 + 16 + laneLo) * lda;
  const bf16* brow0 = B + (size_t)(n0 + laneLo)      * ldb;
  const bf16* brow1 = B + (size_t)(n0 + 16 + laneLo) * ldb;

  for (int kb = 0; kb < K; kb += 32) {
    v16bf a0 = load_a_frag(arow0 + kb, hi);
    v16bf a1 = load_a_frag(arow1 + kb, hi);
    v16bf b0 = load_b_frag(brow0 + kb, hi);
    v16bf b1 = load_b_frag(brow1 + kb, hi);
    acc00 = __builtin_amdgcn_wmma_f32_16x16x32_bf16(false, a0, false, b0, (short)0, acc00, false, false);
    acc01 = __builtin_amdgcn_wmma_f32_16x16x32_bf16(false, a0, false, b1, (short)0, acc01, false, false);
    acc10 = __builtin_amdgcn_wmma_f32_16x16x32_bf16(false, a1, false, b0, (short)0, acc10, false, false);
    acc11 = __builtin_amdgcn_wmma_f32_16x16x32_bf16(false, a1, false, b1, (short)0, acc11, false, false);
  }

  const float bia0 = bias[n0 + laneLo];
  const float bia1 = bias[n0 + 16 + laneLo];

  // C/D layout: VGPR i -> row m0 + hi*8 + i, col n0 + laneLo
  v8f accs[4] = { acc00, acc01, acc10, acc11 };
  #pragma unroll
  for (int ti = 0; ti < 2; ++ti) {
    #pragma unroll
    for (int tj = 0; tj < 2; ++tj) {
      v8f acc = accs[ti * 2 + tj];
      const int ncol = n0 + tj * 16 + laneLo;
      const float bia = tj ? bia1 : bia0;
      #pragma unroll
      for (int i = 0; i < 8; ++i) {
        const int r = m0 + ti * 16 + hi * 8 + i;
        float v = acc[i] + bia;
        if (relu) v = v > 0.f ? v : 0.f;
        if (Cf) Cf[(size_t)r * ldcf + ncol] = v;
        if (Cb) Cb[(size_t)r * ldcb + ncol] = (bf16)v;
      }
    }
  }
}

// ---------------------------------------------------------------------------
// Fused 3-layer conv1d (k=3, pad=1) net, entirely in LDS per sequence.
// reads layout: (S, L, 4). Output y: (S, 64, 128) f32. Layers 1,2 ReLU.
// ---------------------------------------------------------------------------
__global__ __launch_bounds__(256)
void convnet_kernel(const float* __restrict__ reads,
                    const float* __restrict__ W0, const float* __restrict__ b0,
                    const float* __restrict__ W1, const float* __restrict__ b1,
                    const float* __restrict__ W2, const float* __restrict__ b2,
                    float* __restrict__ y)
{
  __shared__ float bufA[CC * LL];
  __shared__ float bufB[CC * LL];
  const int s = blockIdx.x, tid = threadIdx.x;

  for (int idx = tid; idx < 4 * LL; idx += 256) {       // stage (4, L) input
    int ci = idx >> 7, l = idx & 127;
    bufA[idx] = reads[(size_t)s * (LL * 4) + l * 4 + ci];
  }
  __syncthreads();

  for (int o = tid; o < CC * LL; o += 256) {            // layer 1 (Cin=4)
    int co = o >> 7, l = o & 127;
    float acc = b0[co];
    for (int ci = 0; ci < 4; ++ci) {
      #pragma unroll
      for (int kk = 0; kk < 3; ++kk) {
        int ll = l + kk - 1;
        if (ll >= 0 && ll < LL) acc += bufA[ci * LL + ll] * W0[co * 12 + ci * 3 + kk];
      }
    }
    bufB[o] = fmaxf(acc, 0.f);
  }
  __syncthreads();

  for (int o = tid; o < CC * LL; o += 256) {            // layer 2 (Cin=64)
    int co = o >> 7, l = o & 127;
    float acc = b1[co];
    const float* wp = W1 + co * (CC * 3);
    for (int ci = 0; ci < CC; ++ci) {
      #pragma unroll
      for (int kk = 0; kk < 3; ++kk) {
        int ll = l + kk - 1;
        if (ll >= 0 && ll < LL) acc += bufB[ci * LL + ll] * wp[ci * 3 + kk];
      }
    }
    bufA[o] = fmaxf(acc, 0.f);
  }
  __syncthreads();

  for (int o = tid; o < CC * LL; o += 256) {            // layer 3 (linear)
    int co = o >> 7, l = o & 127;
    float acc = b2[co];
    const float* wp = W2 + co * (CC * 3);
    for (int ci = 0; ci < CC; ++ci) {
      #pragma unroll
      for (int kk = 0; kk < 3; ++kk) {
        int ll = l + kk - 1;
        if (ll >= 0 && ll < LL) acc += bufA[ci * LL + ll] * wp[ci * 3 + kk];
      }
    }
    y[(size_t)s * (CC * LL) + o] = acc;
  }
}

// ---------------------------------------------------------------------------
// Window extraction: instance-norm belief window + raw original window -> bf16
// win layout (S, 2*C*W): [c*9+j] = norm belief, [576 + c*9 + j] = original
// ---------------------------------------------------------------------------
__global__ __launch_bounds__(64)
void window_kernel(const float* __restrict__ belief, const float* __restrict__ original,
                   bf16* __restrict__ win, int start)
{
  const int s = blockIdx.x, c = threadIdx.x;
  const float* bp = belief   + (size_t)s * (CC * LL) + c * LL + start;
  const float* op = original + (size_t)s * (CC * LL) + c * LL + start;
  float v[WW], mu = 0.f;
  #pragma unroll
  for (int j = 0; j < WW; ++j) { v[j] = bp[j]; mu += v[j]; }
  mu *= (1.f / WW);
  float var = 0.f;
  #pragma unroll
  for (int j = 0; j < WW; ++j) { float d = v[j] - mu; var += d * d; }
  var *= (1.f / WW);
  const float inv = rsqrtf(var + 1e-5f);
  bf16* w = win + (size_t)s * (2 * CC * WW);
  #pragma unroll
  for (int j = 0; j < WW; ++j) {
    w[c * WW + j]            = (bf16)((v[j] - mu) * inv);
    w[CC * WW + c * WW + j]  = (bf16)op[j];
  }
}

// LSTM pointwise: gates (S,1024) -> c,h; write h (bf16) into zcat[:,256:512]
// (next-step gate GEMM + qkv GEMM input) and ui[:,0:256] (update MLP input).
__global__ __launch_bounds__(256)
void lstm_kernel(const float* __restrict__ gates, float* __restrict__ cst,
                 bf16* __restrict__ zcat, bf16* __restrict__ ui)
{
  const int idx = blockIdx.x * 256 + threadIdx.x;     // S*256
  const int s = idx >> 8, d = idx & 255;
  const float* g = gates + (size_t)s * (4 * DD);
  const float ig = g[d], fg = g[DD + d], gg = g[2 * DD + d], og = g[3 * DD + d];
  const float si = 1.f / (1.f + __expf(-ig));
  const float sf = 1.f / (1.f + __expf(-fg));
  const float so = 1.f / (1.f + __expf(-og));
  const float cn = sf * cst[idx] + si * tanhf(gg);
  const float h  = so * tanhf(cn);
  cst[idx] = cn;
  zcat[(size_t)s * (2 * DD) + DD + d] = (bf16)h;
  ui[(size_t)s * (2 * DD) + d]        = (bf16)h;
}

// Per-(cluster, head) 16x16 attention over D=256 + mean over reads -> bf16
__global__ __launch_bounds__(256)
void attention_kernel(const float* __restrict__ qkv, bf16* __restrict__ heads)
{
  __shared__ float qs[RR * DD], ks[RR * DD], vs[RR * DD];
  __shared__ float sc[RR * RR], csum[RR];
  const int b = blockIdx.x, head = blockIdx.y, tid = threadIdx.x;

  for (int idx = tid; idx < RR * DD; idx += 256) {
    int n = idx >> 8, d = idx & 255;
    const float* row = qkv + (size_t)(b * RR + n) * (3 * NHH * DD) + head * (3 * DD);
    qs[idx] = row[d];
    ks[idx] = row[DD + d];
    vs[idx] = row[2 * DD + d];
  }
  __syncthreads();
  {
    const int n = tid >> 4, m = tid & 15;
    float acc = 0.f;
    for (int d = 0; d < DD; ++d) acc += qs[n * DD + d] * ks[m * DD + d];
    sc[n * RR + m] = acc;
  }
  __syncthreads();
  if (tid < RR) {                       // softmax row tid
    float mx = -1e30f;
    for (int m = 0; m < RR; ++m) mx = fmaxf(mx, sc[tid * RR + m]);
    float sum = 0.f;
    for (int m = 0; m < RR; ++m) { float e = __expf(sc[tid * RR + m] - mx); sc[tid * RR + m] = e; sum += e; }
    const float inv = 1.f / sum;
    for (int m = 0; m < RR; ++m) sc[tid * RR + m] *= inv;
  }
  __syncthreads();
  if (tid < RR) {                       // column sums (mean over n folded in)
    float a = 0.f;
    for (int n = 0; n < RR; ++n) a += sc[n * RR + tid];
    csum[tid] = a;
  }
  __syncthreads();
  {
    const int d = tid;                  // 256 threads = 256 dims
    float acc = 0.f;
    #pragma unroll
    for (int m = 0; m < RR; ++m) acc += csum[m] * vs[m * DD + d];
    heads[(size_t)b * (NHH * DD) + head * DD + d] = (bf16)(acc * (1.f / RR));
  }
}

__global__ __launch_bounds__(256)
void bcast_gf_kernel(const float* __restrict__ gf, bf16* __restrict__ ui)
{
  const int s = blockIdx.x, d = threadIdx.x;
  ui[(size_t)s * (2 * DD) + DD + d] = (bf16)gf[(size_t)(s >> 4) * DD + d];
}

__global__ __launch_bounds__(256)
void belief_update_kernel(float* __restrict__ belief, const float* __restrict__ upd, int start)
{
  const int idx = blockIdx.x * 256 + threadIdx.x;     // S*512
  const int s = idx >> 9, r = idx & 511;
  const int cch = r >> 3, j = r & 7;
  belief[(size_t)s * (CC * LL) + cch * LL + start + 1 + j] += upd[idx];
}

__global__ void pred2_kernel(const float* __restrict__ p1, const float* __restrict__ P1,
                             const float* __restrict__ pb1, float* __restrict__ out,
                             int t, int T)
{
  const int b = blockIdx.x, j = threadIdx.x;
  if (j < 5) {
    float acc = pb1[j];
    const float* x = p1 + (size_t)b * 128;
    const float* w = P1 + (size_t)j * 128;
    for (int k = 0; k < 128; ++k) acc += x[k] * w[k];
    out[(size_t)b * T * 5 + t * 5 + j] = acc;
  }
}

__global__ __launch_bounds__(256)
void zero_state_kernel(float* __restrict__ cst, bf16* __restrict__ zcat)
{
  const int idx = blockIdx.x * 256 + threadIdx.x;     // S*256
  cst[idx] = 0.f;
  const int s = idx >> 8, d = idx & 255;
  zcat[(size_t)s * (2 * DD) + DD + d] = (bf16)0.f;
}

__global__ void cvt_bf16_kernel(const float* __restrict__ src, bf16* __restrict__ dst,
                                int rows, int cols, int dstride)
{
  const int idx = blockIdx.x * 256 + threadIdx.x;
  if (idx >= rows * cols) return;
  const int r = idx / cols, c = idx - r * cols;
  dst[(size_t)r * dstride + c] = (bf16)src[idx];
}

__global__ void addvec_kernel(const float* __restrict__ a, const float* __restrict__ b,
                              float* __restrict__ o, int n)
{
  const int i = blockIdx.x * 256 + threadIdx.x;
  if (i < n) o[i] = a[i] + b[i];
}

// ---------------------------------------------------------------------------
extern "C" void kernel_launch(void* const* d_in, const int* in_sizes, int n_in,
                              void* d_out, int out_size, void* d_ws, size_t ws_size,
                              hipStream_t stream)
{
  (void)in_sizes; (void)n_in; (void)ws_size;
  const int T = out_size / (BC * 5);                   // out_length (=120)

  auto IN = [&](int i) { return (const float*)d_in[i]; };
  const float* reads = IN(0);
  float* outp = (float*)d_out;

  char* ws = (char*)d_ws;
  size_t off = 0;
  auto alloc = [&](size_t bytes) -> char* {
    off = (off + 255) & ~(size_t)255;
    char* p = ws + off; off += bytes; return p;
  };

  float* original = (float*)alloc((size_t)SS * CC * LL * 4);
  float* belief   = (float*)alloc((size_t)SS * CC * LL * 4);
  bf16* wW0b  = (bf16*)alloc((size_t)256 * 1152 * 2);
  bf16* wW1b  = (bf16*)alloc((size_t)256 * 256 * 2);
  bf16* Wcat  = (bf16*)alloc((size_t)1024 * 512 * 2);
  float* bcat = (float*)alloc(1024 * 4);
  bf16* Wqkvb = (bf16*)alloc((size_t)3072 * 256 * 2);
  float* bqkv = (float*)alloc(3072 * 4);
  bf16* Wgb   = (bf16*)alloc((size_t)256 * 1024 * 2);
  bf16* U0b   = (bf16*)alloc((size_t)256 * 512 * 2);
  bf16* U1b   = (bf16*)alloc((size_t)512 * 256 * 2);
  bf16* P0b   = (bf16*)alloc((size_t)128 * 256 * 2);
  bf16* win   = (bf16*)alloc((size_t)SS * 1152 * 2);
  bf16* z1    = (bf16*)alloc((size_t)SS * 256 * 2);
  bf16* zcat  = (bf16*)alloc((size_t)SS * 512 * 2);
  float* gates= (float*)alloc((size_t)SS * 1024 * 4);
  float* cst  = (float*)alloc((size_t)SS * 256 * 4);
  float* qkv  = (float*)alloc((size_t)SS * 3072 * 4);
  bf16* headsb= (bf16*)alloc((size_t)BC * 1024 * 2);
  float* gf   = (float*)alloc((size_t)BC * 256 * 4);
  bf16* gfb   = (bf16*)alloc((size_t)BC * 256 * 2);
  bf16* ui    = (bf16*)alloc((size_t)SS * 512 * 2);
  bf16* u1    = (bf16*)alloc((size_t)SS * 256 * 2);
  float* upd  = (float*)alloc((size_t)SS * 512 * 4);
  float* p1   = (float*)alloc((size_t)128 * 128 * 4);

  auto cvt = [&](const float* src, bf16* dst, int rows, int cols, int dstride) {
    const int n = rows * cols;
    cvt_bf16_kernel<<<(n + 255) / 256, 256, 0, stream>>>(src, dst, rows, cols, dstride);
  };

  // ---- weight prep (bf16 conversion + fusion concats), idempotent -------
  cvt(IN(13), wW0b, 256, 1152, 1152);
  cvt(IN(15), wW1b, 256, 256, 256);
  cvt(IN(17), Wcat,       1024, 256, 512);   // Wih -> cols [0,256)
  cvt(IN(19), Wcat + 256, 1024, 256, 512);   // Whh -> cols [256,512)
  addvec_kernel<<<4, 256, 0, stream>>>(IN(18), IN(20), bcat, 1024);
  for (int h = 0; h < NHH; ++h)
    for (int p = 0; p < 3; ++p) {
      cvt(IN(21 + h * 6 + p * 2), Wqkvb + (size_t)(h * 768 + p * 256) * 256, 256, 256, 256);
      hipMemcpyAsync(bqkv + h * 768 + p * 256, d_in[21 + h * 6 + p * 2 + 1],
                     256 * sizeof(float), hipMemcpyDeviceToDevice, stream);
    }
  cvt(IN(45), Wgb, 256, 1024, 1024);
  cvt(IN(47), U0b, 256, 512, 512);
  cvt(IN(49), U1b, 512, 256, 256);
  cvt(IN(51), P0b, 128, 256, 256);

  // ---- conv nets (fused in-LDS) + state init ----------------------------
  convnet_kernel<<<SS, 256, 0, stream>>>(reads, IN(1), IN(2), IN(3), IN(4), IN(5), IN(6), original);
  convnet_kernel<<<SS, 256, 0, stream>>>(reads, IN(7), IN(8), IN(9), IN(10), IN(11), IN(12), belief);
  zero_state_kernel<<<SS, 256, 0, stream>>>(cst, zcat);

  auto gemm = [&](const bf16* A, int lda, const bf16* B, int ldb, const float* bias,
                  float* Cf, int ldcf, bf16* Cb, int ldcb, int M, int N, int K, int relu) {
    dim3 grid(N / 64, M / 128);
    gemm_bf16_wmma<<<grid, 256, 0, stream>>>(A, lda, B, ldb, bias, Cf, ldcf, Cb, ldcb, K, relu);
  };

  // ---- 120-step recurrent scan ------------------------------------------
  for (int t = 0; t < T; ++t) {
    window_kernel<<<SS, 64, 0, stream>>>(belief, original, win, t);
    // window MLP: 1152 -> 256 (relu) -> 256, z goes into zcat[:,0:256]
    gemm(win, 1152, wW0b, 1152, IN(14), nullptr, 0, z1, 256, SS, 256, 1152, 1);
    gemm(z1, 256, wW1b, 256, IN(16), nullptr, 0, zcat, 512, SS, 256, 256, 0);
    // fused LSTM gates: [z,h] x [Wih;Whh]^T + (bih+bhh)
    gemm(zcat, 512, Wcat, 512, bcat, gates, 1024, nullptr, 0, SS, 1024, 512, 0);
    lstm_kernel<<<SS, 256, 0, stream>>>(gates, cst, zcat, ui);
    // all-head QKV in one GEMM from h (strided view of zcat)
    gemm(zcat + DD, 512, Wqkvb, 256, bqkv, qkv, 3072, nullptr, 0, SS, 3072, 256, 0);
    attention_kernel<<<dim3(BC, NHH), 256, 0, stream>>>(qkv, headsb);
    // glin: (128,1024)->(128,256) relu, f32 + bf16 outputs
    gemm(headsb, 1024, Wgb, 1024, IN(46), gf, 256, gfb, 256, BC, 256, 1024, 1);
    bcast_gf_kernel<<<SS, 256, 0, stream>>>(gf, ui);
    // update MLP: 512 -> 256 (relu) -> 512
    gemm(ui, 512, U0b, 512, IN(48), nullptr, 0, u1, 256, SS, 256, 512, 1);
    gemm(u1, 256, U1b, 256, IN(50), upd, 512, nullptr, 0, SS, 512, 256, 0);
    belief_update_kernel<<<SS * 2, 256, 0, stream>>>(belief, upd, t);
    // pred head: 256 -> 128 (relu, WMMA) -> 5 (tiny, VALU)
    gemm(gfb, 256, P0b, 256, IN(52), p1, 128, nullptr, 0, BC, 128, 256, 1);
    pred2_kernel<<<BC, 32, 0, stream>>>(p1, IN(53), IN(54), outp, t, T);
  }
}
